// MoE_11922829214184
// MI455X (gfx1250) — matrix operand order repaired and checked by reading, not compiled, not run on hardware
//
#include <hip/hip_runtime.h>
#include <hip/hip_bf16.h>

#define Bb 2
#define Tt 1024
#define Cc 1024
#define Ii 4096
#define Ee 8
#define NTOK (Bb * Tt)            // 2048 tokens
#define RH (Cc / 4)               // 256 router hidden
#define CHUNK 128                 // I-chunk per iteration
#define MAXPE 2048                // max tokens per expert
#define TILE_M 32                 // tokens per FFN block (2 WMMA M-tiles)
#define TILES_PER_E (MAXPE / TILE_M)  // 64

typedef __attribute__((ext_vector_type(16))) __bf16 v16bf;
typedef __attribute__((ext_vector_type(8)))  __bf16 v8bf;
typedef __attribute__((ext_vector_type(8)))  float  v8f;

__device__ __forceinline__ __bf16 f2bf(float f) {
  unsigned u = __builtin_bit_cast(unsigned, f);
  u += 0x7FFFu + ((u >> 16) & 1u);
  unsigned short h = (unsigned short)(u >> 16);
  return __builtin_bit_cast(__bf16, h);
}

__device__ __forceinline__ v16bf cat16(v8bf lo, v8bf hi) {
  return __builtin_shufflevector(lo, hi, 0, 1, 2, 3, 4, 5, 6, 7,
                                 8, 9, 10, 11, 12, 13, 14, 15);
}

// ---------------------------------------------------------------------------
// Kernel 0: zero output + expert counters
// ---------------------------------------------------------------------------
__global__ void moe_zero_kernel(float* __restrict__ out, int n, int* __restrict__ counts) {
  int i = blockIdx.x * blockDim.x + threadIdx.x;
  if (i < n) out[i] = 0.f;
  if (blockIdx.x == 0 && threadIdx.x < Ee) counts[threadIdx.x] = 0;
}

// ---------------------------------------------------------------------------
// Kernel 1: convert x fp32 -> bf16 (row-major), 4 elements per thread
// ---------------------------------------------------------------------------
struct bf4 { __bf16 a, b, c, d; };
__global__ void moe_convx_kernel(const float* __restrict__ src, __bf16* __restrict__ dst, int n4) {
  int i = blockIdx.x * blockDim.x + threadIdx.x;
  if (i < n4) {
    float4 v = ((const float4*)src)[i];
    bf4 o = { f2bf(v.x), f2bf(v.y), f2bf(v.z), f2bf(v.w) };
    ((bf4*)dst)[i] = o;
  }
}

// ---------------------------------------------------------------------------
// Kernel 2: convert + swizzle weights into WMMA B-operand tile order.
//   src: nMat matrices of [K, N] fp32 (row-major)
//   dst: [mat][nTile=N/16][kTile=K/32][lane=32][16] bf16
//   Register element i of lane L = src[kTile*32 + 8*(L>>4) + i + (i>=8 ? 8 : 0)][nTile*16 + (L&15)]
// ---------------------------------------------------------------------------
__global__ void moe_convw_kernel(const float* __restrict__ src, __bf16* __restrict__ dst,
                                 int K, int N, int nMat) {
  long gid = (long)blockIdx.x * blockDim.x + threadIdx.x;
  int lanesPerMat = (K / 32) * (N / 16) * 32;
  if (gid >= (long)nMat * lanesPerMat) return;
  int  m    = (int)(gid / lanesPerMat);
  long r    = gid % lanesPerMat;
  int  lane = (int)(r & 31);
  long tIdx = r >> 5;
  int  kTile = (int)(tIdx % (K / 32));
  long nTile = tIdx / (K / 32);
  int  col  = (int)nTile * 16 + (lane & 15);
  int  row0 = kTile * 32 + 8 * (lane >> 4);
  const float* s = src + (size_t)m * K * N + (size_t)row0 * N + col;
  __bf16* d = dst + ((((size_t)m * (N / 16) + nTile) * (K / 32) + kTile) * 32 + lane) * 16;
#pragma unroll
  for (int i = 0; i < 16; ++i) d[i] = f2bf(s[(size_t)(i + (i >= 8 ? 8 : 0)) * N]);
}

// ---------------------------------------------------------------------------
// Kernel 3: router — one block (256 threads) per token.
// ---------------------------------------------------------------------------
__global__ void moe_router_kernel(const float* __restrict__ x,
                                  const float* __restrict__ rw1,
                                  const float* __restrict__ rb1,
                                  const float* __restrict__ rw2,
                                  const float* __restrict__ rb2,
                                  int* __restrict__ counts,
                                  int* __restrict__ tokList,
                                  float* __restrict__ gateList) {
  const int n = blockIdx.x;
  const int t = threadIdx.x;
  __shared__ float hid[RH];
  __shared__ float lg[Ee];

  const float* xr = x + (size_t)n * Cc;
  float acc = rb1[t];
  for (int c = 0; c < Cc; ++c) acc = fmaf(xr[c], rw1[(size_t)c * RH + t], acc);
  hid[t] = fmaxf(acc, 0.f);
  __syncthreads();

  if (t < Ee) {
    float l = rb2[t];
    for (int k = 0; k < RH; ++k) l = fmaf(hid[k], rw2[(size_t)k * Ee + t], l);
    lg[t] = l;
  }
  __syncthreads();

  if (t == 0) {
    float m = lg[0];
    for (int e = 1; e < Ee; ++e) m = fmaxf(m, lg[e]);
    float g[Ee], s = 0.f;
    for (int e = 0; e < Ee; ++e) { g[e] = __expf(lg[e] - m); s += g[e]; }
    float inv = 1.f / s;
    for (int e = 0; e < Ee; ++e) g[e] *= inv;
    int i1 = 0;
    for (int e = 1; e < Ee; ++e) if (g[e] > g[i1]) i1 = e;
    int i2 = (i1 == 0) ? 1 : 0;
    for (int e = 0; e < Ee; ++e) if (e != i1 && g[e] > g[i2]) i2 = e;
    float e1 = __expf(g[i1] * 0.5f), e2 = __expf(g[i2] * 0.5f);
    float w1 = e1 / (e1 + e2), w2 = e2 / (e1 + e2);
    int p1 = atomicAdd(&counts[i1], 1);
    tokList[i1 * MAXPE + p1]  = n;
    gateList[i1 * MAXPE + p1] = w1;
    int p2 = atomicAdd(&counts[i2], 1);
    tokList[i2 * MAXPE + p2]  = n;
    gateList[i2 * MAXPE + p2] = w2;
  }
}

// ---------------------------------------------------------------------------
// Kernel 4: fused expert FFN for a 32-token tile of one expert.
//   blockIdx.y = expert (0..7 routed, 8 = shared expert), blockIdx.x = tile.
//   8 wave32s; bf16 WMMA 16x16x32, f32 accumulate; pre-swizzled bf16 weights
//   fetched as 2x global_load_b128 per B operand, shared by TWO M-tiles
//   (2 WMMAs per B fetch); software-pipelined one k-step ahead; x gathered
//   into LDS via async global->LDS b128 copies.
// ---------------------------------------------------------------------------
__global__ __launch_bounds__(256)
void moe_ffn_kernel(const __bf16* __restrict__ xbf,
                    const __bf16* __restrict__ w1s,  // [9][I/16][C/32][32][16]
                    const __bf16* __restrict__ w2s,  // [9][C/16][I/32][32][16]
                    const float* __restrict__ eb1, const float* __restrict__ eb2,
                    const float* __restrict__ sb1, const float* __restrict__ sb2,
                    const int* __restrict__ counts,
                    const int* __restrict__ tokList,
                    const float* __restrict__ gateList,
                    float* __restrict__ out) {
  const int e    = blockIdx.y;   // 0..8
  const int tile = blockIdx.x;   // 0..63
  const int tid  = threadIdx.x;
  const int wv   = tid >> 5;     // 8 waves
  const int lane = tid & 31;
  const int half = lane >> 4;
  const int nn   = lane & 15;

  int cnt;
  const float *b1, *b2;
  if (e < Ee) {
    cnt = counts[e];
    if (tile * TILE_M >= cnt) return;        // uniform early-out
    b1 = eb1 + (size_t)e * Ii;  b2 = eb2 + (size_t)e * Cc;
  } else {
    cnt = NTOK;
    b1 = sb1;  b2 = sb2;
  }
  const __bf16* w1m = w1s + (size_t)e * Cc * Ii;
  const __bf16* w2m = w2s + (size_t)e * Ii * Cc;

  __shared__ __align__(16) __bf16 xA[TILE_M * Cc];     // 64 KB gathered x tile
  __shared__ __align__(16) __bf16 hS[TILE_M * CHUNK];  //  8 KB silu(h) chunk
  __shared__ int   tokL[TILE_M];
  __shared__ float gateL[TILE_M];

  if (tid < TILE_M) {
    int r = tile * TILE_M + tid;
    if (r < cnt) {
      if (e < Ee) { tokL[tid] = tokList[e * MAXPE + r]; gateL[tid] = gateList[e * MAXPE + r]; }
      else        { tokL[tid] = r;                      gateL[tid] = 1.f; }
    } else        { tokL[tid] = 0;                      gateL[tid] = 0.f; }
  }
  __syncthreads();

  // ---- gather x tile into LDS with async global->LDS copies (16B each).
  // Padded rows point at token 0; their gate is 0 so they contribute nothing.
  for (int idx = tid; idx < (TILE_M * Cc) / 8; idx += 256) {
    int r  = idx >> 7;          // 128 x 16B per row
    int c8 = idx & 127;
    unsigned long long ga =
        (unsigned long long)(uintptr_t)(xbf + (size_t)tokL[r] * Cc + c8 * 8);
    unsigned lo = (unsigned)(uintptr_t)(&xA[r * Cc + c8 * 8]);
    asm volatile("global_load_async_to_lds_b128 %0, %1, off"
                 :: "v"(lo), "v"(ga) : "memory");
  }
  asm volatile("s_wait_asynccnt 0x0" ::: "memory");
  __syncthreads();

  v8f zero = {};
  v8f outAcc[2][8];              // [m-tile][n-tile]; wave wv owns cols [wv*128,+128)
#pragma unroll
  for (int m = 0; m < 2; ++m)
#pragma unroll
    for (int t = 0; t < 8; ++t) outAcc[m][t] = zero;

  const int kT1 = Cc / 32;       // k-tiles in w1s
  const int kT2 = Ii / 32;       // k-tiles in w2s

  for (int i0 = 0; i0 < Ii; i0 += CHUNK) {
    if (lane == 0 && i0 + CHUNK < Ii) {
      __builtin_prefetch(w1m + (((size_t)((i0 + CHUNK) >> 4) + wv) * kT1) * 512, 0, 1);
      __builtin_prefetch(w2m + (((size_t)(wv * 8)) * kT2 + ((i0 + CHUNK) >> 5)) * 512, 0, 1);
    }
    // ------- GEMM1: h[32,CHUNK] = silu(x @ W1[:, i0:i0+CHUNK] + b1)
    auto loadB1 = [&](int kk) -> v16bf {
      const v8bf* bp = (const v8bf*)
          (w1m + ((((size_t)(i0 >> 4) + wv) * kT1 + (kk >> 5)) * 32 + lane) * 16);
      return cat16(bp[0], bp[1]);
    };
    auto loadA1 = [&](int kk, int m) -> v16bf {
      const v8bf* ap = (const v8bf*)&xA[(m * 16 + nn) * Cc + kk + 8 * half];
      return cat16(ap[0], ap[2]);
    };
    v8f acc0 = zero, acc1 = zero;
    v16bf b  = loadB1(0);
    v16bf a0 = loadA1(0, 0);
    v16bf a1 = loadA1(0, 1);
    for (int kk = 0; kk < Cc; kk += 32) {
      v16bf bn = b, a0n = a0, a1n = a1;
      if (kk + 32 < Cc) {                       // prefetch next k-step
        bn  = loadB1(kk + 32);
        a0n = loadA1(kk + 32, 0);
        a1n = loadA1(kk + 32, 1);
      }
      acc0 = __builtin_amdgcn_wmma_f32_16x16x32_bf16(false, a0, false, b,
                                                     (short)0, acc0, false, false);
      acc1 = __builtin_amdgcn_wmma_f32_16x16x32_bf16(false, a1, false, b,
                                                     (short)0, acc1, false, false);
      b = bn; a0 = a0n; a1 = a1n;
    }
    // SiLU + bias; wave wv owns h columns [wv*16, wv*16+16), all 32 rows
    {
      float bias = b1[i0 + wv * 16 + nn];
#pragma unroll
      for (int r = 0; r < 8; ++r) {
        int M0 = r + 8 * half;
        float v0 = acc0[r] + bias;
        v0 = v0 / (1.f + __expf(-v0));
        hS[M0 * CHUNK + wv * 16 + nn] = f2bf(v0);
        float v1 = acc1[r] + bias;
        v1 = v1 / (1.f + __expf(-v1));
        hS[(16 + M0) * CHUNK + wv * 16 + nn] = f2bf(v1);
      }
    }
    __syncthreads();
    // ------- GEMM2: out[32,1024] += h @ W2[i0:i0+CHUNK, :]
    auto loadB2 = [&](int kk, int t) -> v16bf {
      const v8bf* bp = (const v8bf*)
          (w2m + ((((size_t)(wv * 8 + t)) * kT2 + ((i0 + kk) >> 5)) * 32 + lane) * 16);
      return cat16(bp[0], bp[1]);
    };
    for (int kk = 0; kk < CHUNK; kk += 32) {
      const v8bf* ap0 = (const v8bf*)&hS[nn * CHUNK + kk + 8 * half];
      v16bf ha0 = cat16(ap0[0], ap0[2]);
      const v8bf* ap1 = (const v8bf*)&hS[(16 + nn) * CHUNK + kk + 8 * half];
      v16bf ha1 = cat16(ap1[0], ap1[2]);
      v16bf bt = loadB2(kk, 0);
#pragma unroll
      for (int t = 0; t < 8; ++t) {
        v16bf btn = (t < 7) ? loadB2(kk, t + 1) : bt;   // prefetch next n-tile
        outAcc[0][t] = __builtin_amdgcn_wmma_f32_16x16x32_bf16(
            false, ha0, false, bt, (short)0, outAcc[0][t], false, false);
        outAcc[1][t] = __builtin_amdgcn_wmma_f32_16x16x32_bf16(
            false, ha1, false, bt, (short)0, outAcc[1][t], false, false);
        bt = btn;
      }
    }
    __syncthreads();   // before hS is rewritten next chunk
  }

  // epilogue: out[token, col] += gate * (acc + b2[col])
#pragma unroll
  for (int t = 0; t < 8; ++t) {
    int col = wv * 128 + t * 16 + nn;
    float bias = b2[col];
#pragma unroll
    for (int m = 0; m < 2; ++m) {
#pragma unroll
      for (int r = 0; r < 8; ++r) {
        int M = m * 16 + r + 8 * half;
        float g = gateL[M];
        if (g != 0.f)
          atomicAdd(&out[(size_t)tokL[M] * Cc + col], g * (outAcc[m][t][r] + bias));
      }
    }
  }
}

// ---------------------------------------------------------------------------
extern "C" void kernel_launch(void* const* d_in, const int* in_sizes, int n_in,
                              void* d_out, int out_size, void* d_ws, size_t ws_size,
                              hipStream_t stream) {
  (void)in_sizes; (void)n_in; (void)out_size; (void)ws_size;
  const float* x   = (const float*)d_in[0];
  const float* rw1 = (const float*)d_in[1];
  const float* rb1 = (const float*)d_in[2];
  const float* rw2 = (const float*)d_in[3];
  const float* rb2 = (const float*)d_in[4];
  const float* ew1 = (const float*)d_in[5];
  const float* eb1 = (const float*)d_in[6];
  const float* ew2 = (const float*)d_in[7];
  const float* eb2 = (const float*)d_in[8];
  const float* sw1 = (const float*)d_in[9];
  const float* sb1 = (const float*)d_in[10];
  const float* sw2 = (const float*)d_in[11];
  const float* sb2 = (const float*)d_in[12];
  float* out = (float*)d_out;

  // workspace layout (bytes):
  //   counts[16] i32 | tokList[8*2048] i32 | gateList[8*2048] f32 |
  //   xbf 4MB bf16 | w1s 72MB bf16 | w2s 72MB bf16
  char* ws = (char*)d_ws;
  int*    counts   = (int*)ws;
  int*    tokList  = counts + 16;
  float*  gateList = (float*)(tokList + Ee * MAXPE);
  __bf16* xbf      = (__bf16*)(ws + (1 << 20));
  __bf16* w1s      = xbf + (size_t)NTOK * Cc;
  __bf16* w2s      = w1s + (size_t)(Ee + 1) * Cc * Ii;

  const int total = NTOK * Cc;
  moe_zero_kernel<<<(total + 255) / 256, 256, 0, stream>>>(out, total, counts);

  // one-time per-launch bf16 conversion + WMMA swizzle of weights and x
  moe_convx_kernel<<<(total / 4 + 255) / 256, 256, 0, stream>>>(x, xbf, total / 4);
  {
    long lanes1 = (long)Ee * (Cc / 32) * (Ii / 16) * 32;   // ew1
    moe_convw_kernel<<<(int)((lanes1 + 255) / 256), 256, 0, stream>>>(
        ew1, w1s, Cc, Ii, Ee);
    long lanes1s = (long)(Cc / 32) * (Ii / 16) * 32;       // sw1
    moe_convw_kernel<<<(int)((lanes1s + 255) / 256), 256, 0, stream>>>(
        sw1, w1s + (size_t)Ee * Cc * Ii, Cc, Ii, 1);
    long lanes2 = (long)Ee * (Ii / 32) * (Cc / 16) * 32;   // ew2
    moe_convw_kernel<<<(int)((lanes2 + 255) / 256), 256, 0, stream>>>(
        ew2, w2s, Ii, Cc, Ee);
    long lanes2s = (long)(Ii / 32) * (Cc / 16) * 32;       // sw2
    moe_convw_kernel<<<(int)((lanes2s + 255) / 256), 256, 0, stream>>>(
        sw2, w2s + (size_t)Ee * Ii * Cc, Ii, Cc, 1);
  }

  moe_router_kernel<<<NTOK, RH, 0, stream>>>(x, rw1, rb1, rw2, rb2,
                                             counts, tokList, gateList);

  dim3 grid(TILES_PER_E, Ee + 1);
  moe_ffn_kernel<<<grid, 256, 0, stream>>>(xbf, w1s, w2s, eb1, eb2, sb1, sb2,
                                           counts, tokList, gateList, out);
}
